// ConstrainedAttention_4415226380584
// MI455X (gfx1250) — compile-verified, matrix operand order checked by reading, not run on hardware
//
#include <hip/hip_runtime.h>
#include <hip/hip_bf16.h>

#define B_    2
#define S_    2048
#define D_    512
#define H_    8
#define HD_   64
#define DSEM_ 256

typedef __bf16 bf16;
typedef __attribute__((ext_vector_type(16))) __bf16 v16bf;
typedef __attribute__((ext_vector_type(8)))  float  v8f;
typedef unsigned int v4u __attribute__((ext_vector_type(4)));

union FragU { v4u u[2]; v16bf v; };
union Pack8 { v4u u; bf16 a[8]; };

// Load a 16x32 bf16 A-fragment (or B^T-fragment) from a row-major matrix.
// Lane layout (ISA 16-bit A 16x32): row = base_row + lane%16, half = lane/16;
// elements 0..7  <- col [colbase + half*8      .. +7]
// elements 8..15 <- col [colbase + 16 + half*8 .. +7]
__device__ __forceinline__ v16bf load_frag(const bf16* base, int ld, int row,
                                           int colbase, int hh) {
  const bf16* p = base + (size_t)row * ld + colbase + hh * 8;
  FragU f;
  f.u[0] = *(const v4u*)(p);
  f.u[1] = *(const v4u*)(p + 16);
  return f.v;
}

__device__ __forceinline__ v8f wmma_bf16(v16bf a, v16bf b, v8f c) {
  return __builtin_amdgcn_wmma_f32_16x16x32_bf16(false, a, false, b, (short)0, c,
                                                 false, false);
}

__device__ __forceinline__ v8f vzero8() {
  v8f c = {0.f, 0.f, 0.f, 0.f, 0.f, 0.f, 0.f, 0.f};
  return c;
}

// ---------------- elementwise f32 -> bf16 ----------------
__global__ void k_convert(const float* __restrict__ src, bf16* __restrict__ dst, int n) {
  int i = blockIdx.x * blockDim.x + threadIdx.x;
  int stride = gridDim.x * blockDim.x;
  for (; i < n; i += stride) dst[i] = (bf16)src[i];
}

// ------------- row-normalize semantic features -> bf16 (one wave / row) -------------
__global__ void k_normalize(const float* __restrict__ sem, bf16* __restrict__ dst) {
  int wave = threadIdx.x >> 5, lane = threadIdx.x & 31;
  int row = blockIdx.x * 4 + wave;  // 0..B*S-1
  const float* src = sem + (size_t)row * DSEM_;
  float v[8];
  float ss = 0.f;
#pragma unroll
  for (int i = 0; i < 8; ++i) { v[i] = src[lane * 8 + i]; ss += v[i] * v[i]; }
#pragma unroll
  for (int m = 1; m <= 16; m <<= 1) ss += __shfl_xor(ss, m, 32);
  float sc = 1.f / fmaxf(sqrtf(ss), 1e-12f);
  bf16* d = dst + (size_t)row * DSEM_;
#pragma unroll
  for (int i = 0; i < 8; ++i) d[lane * 8 + i] = (bf16)(v[i] * sc);
}

// ------------- mean over sequence of query -> [B, D] -------------
__global__ void k_qmean(const float* __restrict__ q, float* __restrict__ qm) {
  int g = blockIdx.x * blockDim.x + threadIdx.x;  // 0..B*D-1
  int b = g >> 9, d = g & (D_ - 1);
  float s = 0.f;
  for (int t = 0; t < S_; ++t) s += q[((size_t)b * S_ + t) * D_ + d];
  qm[g] = s * (1.f / (float)S_);
}

// ------------- temperature MLP -------------
__global__ void k_temp(const float* __restrict__ qm, const float* __restrict__ Wt1,
                       const float* __restrict__ bt1, const float* __restrict__ Wt2,
                       const float* __restrict__ bt2, float* __restrict__ invtemp) {
  __shared__ float red[256];
  int b = blockIdx.x, j = threadIdx.x;
  float a = bt1[j];
  for (int k = 0; k < D_; ++k) a += qm[b * D_ + k] * Wt1[j * D_ + k];
  red[j] = fmaxf(a, 0.f) * Wt2[j];
  __syncthreads();
  for (int s = 128; s > 0; s >>= 1) {
    if (j < s) red[j] += red[j + s];
    __syncthreads();
  }
  if (j == 0) {
    float z = red[0] + bt2[0];
    float t = 1.f / (1.f + __expf(-z));
    invtemp[b] = 1.f / (0.5f + 1.5f * t);
  }
}

// ------------- head-independent semantic bias, computed ONCE -------------
// biasT[b][k][q] = min(sim(q,k) - 0.5, 0) in bf16, k-major so attention reads
// one 16B vector per lane (8 consecutive q for this lane's key).
__global__ void k_simbias(const bf16* __restrict__ Sn, bf16* __restrict__ biasT) {
  int wave = threadIdx.x >> 5, lane = threadIdx.x & 31;
  int hh = lane >> 4, r = lane & 15;
  int kt = blockIdx.x * 4 + wave;  // 0..127
  int qt = blockIdx.y;             // 0..127
  int b = blockIdx.z;
  int q0 = qt * 16, k0 = kt * 16;
  const bf16* Snb = Sn + ((size_t)b * S_) * DSEM_;
  v8f sim = vzero8();
#pragma unroll
  for (int j = 0; j < 8; ++j) {
    v16bf aq = load_frag(Snb, DSEM_, q0 + r, j * 32, hh);
    v16bf bk = load_frag(Snb, DSEM_, k0 + r, j * 32, hh);
    sim = wmma_bf16(aq, bk, sim);
  }
  Pack8 pk;
#pragma unroll
  for (int v = 0; v < 8; ++v) pk.a[v] = (bf16)fminf(sim[v] - 0.5f, 0.f);
  // C-frag: value v <-> (q = q0 + 8*hh + v, key = k0 + r): one aligned 16B store
  *(v4u*)(biasT + ((size_t)b * S_ + k0 + r) * S_ + q0 + 8 * hh) = pk.u;
}

// ------------- WMMA GEMM, 32x64 tile per wave -------------
// out[m,n] = sum_k A[m,k]*W[n,k] + bias[n]
// mode 0/1: -> bf16 [B,H,S,HD]   mode 2: -> bf16 [B,H,HD,S]   mode 3: -> f32 [M,D]
__global__ void k_gemm(const bf16* __restrict__ A, const bf16* __restrict__ W,
                       const float* __restrict__ bias, bf16* __restrict__ outb,
                       float* __restrict__ outf, int mode) {
  int wave = threadIdx.x >> 5, lane = threadIdx.x & 31;
  int hh = lane >> 4, r = lane & 15;
  int mt = blockIdx.x * 4 + wave;  // 0..127 (32 rows each)
  int nt = blockIdx.y;             // 0..7   (64 cols each)
  v8f c[2][4];
#pragma unroll
  for (int mi = 0; mi < 2; ++mi)
#pragma unroll
    for (int f = 0; f < 4; ++f) c[mi][f] = vzero8();
  int ar0 = mt * 32 + r;
  int br0 = nt * 64 + r;
#pragma unroll 2
  for (int kt = 0; kt < D_ / 32; ++kt) {
    int kc = kt * 32;
    __builtin_prefetch(A + (size_t)ar0 * D_ + kc + 32, 0, 0);
    __builtin_prefetch(W + (size_t)br0 * D_ + kc + 32, 0, 0);
    v16bf a0 = load_frag(A, D_, ar0, kc, hh);
    v16bf a1 = load_frag(A, D_, ar0 + 16, kc, hh);
    v16bf b0 = load_frag(W, D_, br0, kc, hh);
    v16bf b1 = load_frag(W, D_, br0 + 16, kc, hh);
    v16bf b2 = load_frag(W, D_, br0 + 32, kc, hh);
    v16bf b3 = load_frag(W, D_, br0 + 48, kc, hh);
    c[0][0] = wmma_bf16(a0, b0, c[0][0]);
    c[0][1] = wmma_bf16(a0, b1, c[0][1]);
    c[0][2] = wmma_bf16(a0, b2, c[0][2]);
    c[0][3] = wmma_bf16(a0, b3, c[0][3]);
    c[1][0] = wmma_bf16(a1, b0, c[1][0]);
    c[1][1] = wmma_bf16(a1, b1, c[1][1]);
    c[1][2] = wmma_bf16(a1, b2, c[1][2]);
    c[1][3] = wmma_bf16(a1, b3, c[1][3]);
  }
#pragma unroll
  for (int mi = 0; mi < 2; ++mi) {
#pragma unroll
    for (int f = 0; f < 4; ++f) {
      int n = nt * 64 + f * 16 + r;
      float bn = bias[n];
#pragma unroll
      for (int v = 0; v < 8; ++v) {
        int m = mt * 32 + mi * 16 + v + 8 * hh;
        float val = c[mi][f][v] + bn;
        if (mode == 3) {
          outf[(size_t)m * D_ + n] = val;
        } else {
          int bb = m >> 11, s = m & (S_ - 1);
          int h = n >> 6, hd = n & 63;
          size_t idx = (mode == 2)
                           ? ((((size_t)bb * H_ + h) * HD_ + hd) * S_ + s)
                           : ((((size_t)bb * H_ + h) * S_ + s) * HD_ + hd);
          outb[idx] = (bf16)val;
        }
      }
    }
  }
}

// ------------- fused flash attention (bias precomputed) -------------
__global__ void k_attn(const bf16* __restrict__ Qh, const bf16* __restrict__ Kh,
                       const bf16* __restrict__ Vt, const bf16* __restrict__ biasT,
                       const int* __restrict__ mask, const float* __restrict__ invtemp,
                       bf16* __restrict__ ctx) {
  __shared__ alignas(16) bf16 lds_p[4][16 * 32];
  int wave = threadIdx.x >> 5, lane = threadIdx.x & 31;
  int hh = lane >> 4, r = lane & 15;
  int h = blockIdx.y, b = blockIdx.z;
  int q0 = (blockIdx.x * 4 + wave) * 16;

  const bf16* Qb = Qh + (((size_t)b * H_ + h) * S_) * HD_;
  const bf16* Kb = Kh + (((size_t)b * H_ + h) * S_) * HD_;
  const bf16* Vb = Vt + (((size_t)b * H_ + h) * HD_) * S_;
  const bf16* Bb = biasT + ((size_t)b * S_) * S_;  // [k][q]
  const int* mk = mask + b * S_;
  float sscale = 0.125f * invtemp[b];  // 1/sqrt(64) / temperature

  v16bf qa[2];
#pragma unroll
  for (int kk = 0; kk < 2; ++kk) qa[kk] = load_frag(Qb, HD_, q0 + r, kk * 32, hh);

  v8f acc[4];
#pragma unroll
  for (int f = 0; f < 4; ++f) acc[f] = vzero8();
  float mrow[8], lrow[8];
#pragma unroll
  for (int v = 0; v < 8; ++v) { mrow[v] = -1e30f; lrow[v] = 0.f; }

  for (int kb = 0; kb < S_; kb += 32) {
    __builtin_prefetch(Kb + (size_t)(kb + 32 + r) * HD_, 0, 0);
    float st[2][8];
#pragma unroll
    for (int ntile = 0; ntile < 2; ++ntile) {
      int key = kb + ntile * 16 + r;  // this lane's key column in the C-frag
      v8f s = vzero8();
#pragma unroll
      for (int kk = 0; kk < 2; ++kk)
        s = wmma_bf16(qa[kk], load_frag(Kb, HD_, key, kk * 32, hh), s);
      // precomputed semantic bias: 8 consecutive q for this lane's key
      Pack8 pk;
      pk.u = *(const v4u*)(Bb + (size_t)key * S_ + q0 + 8 * hh);
      int mval = mk[key];
#pragma unroll
      for (int v = 0; v < 8; ++v) {
        int qrow = q0 + v + 8 * hh;
        float diff = fabsf((float)(qrow - key));
        float pb = -0.1f * fmaxf(diff - 10.f, 0.f);  // position bias
        float sv = s[v] * sscale + (float)pk.a[v] + pb;
        st[ntile][v] = (mval == 0) ? -1e30f : sv;
      }
    }
    // online softmax: per-row max/sum across the 16 lanes of each wave half
    float alpha[8];
#pragma unroll
    for (int v = 0; v < 8; ++v) {
      float t = fmaxf(st[0][v], st[1][v]);
      t = fmaxf(t, __shfl_xor(t, 1, 32));
      t = fmaxf(t, __shfl_xor(t, 2, 32));
      t = fmaxf(t, __shfl_xor(t, 4, 32));
      t = fmaxf(t, __shfl_xor(t, 8, 32));
      float nm = fmaxf(mrow[v], t);
      alpha[v] = __expf(mrow[v] - nm);
      mrow[v] = nm;
      float p0 = __expf(st[0][v] - nm);
      float p1 = __expf(st[1][v] - nm);
      st[0][v] = p0;
      st[1][v] = p1;
      float rs = p0 + p1;
      rs += __shfl_xor(rs, 1, 32);
      rs += __shfl_xor(rs, 2, 32);
      rs += __shfl_xor(rs, 4, 32);
      rs += __shfl_xor(rs, 8, 32);
      lrow[v] = lrow[v] * alpha[v] + rs;
    }
#pragma unroll
    for (int f = 0; f < 4; ++f)
#pragma unroll
      for (int v = 0; v < 8; ++v) acc[f][v] *= alpha[v];

    // C-layout -> A-layout transpose of P via LDS, converting to bf16
    __syncthreads();
#pragma unroll
    for (int ntile = 0; ntile < 2; ++ntile)
#pragma unroll
      for (int v = 0; v < 8; ++v)
        lds_p[wave][(v + 8 * hh) * 32 + ntile * 16 + r] = (bf16)st[ntile][v];
    __syncthreads();
    v16bf pa = load_frag(&lds_p[wave][0], 32, r, 0, hh);

    // PV: B^T row = hd (Vt is [B,H,HD,S], contiguous along keys)
#pragma unroll
    for (int f = 0; f < 4; ++f) {
      v16bf vb = load_frag(Vb, S_, f * 16 + r, kb, hh);
      acc[f] = wmma_bf16(pa, vb, acc[f]);
    }
  }

#pragma unroll
  for (int f = 0; f < 4; ++f) {
    int hd = h * HD_ + f * 16 + r;
#pragma unroll
    for (int v = 0; v < 8; ++v) {
      int qrow = q0 + v + 8 * hh;
      ctx[((size_t)b * S_ + qrow) * D_ + hd] = (bf16)(acc[f][v] / lrow[v]);
    }
  }
}

extern "C" void kernel_launch(void* const* d_in, const int* in_sizes, int n_in,
                              void* d_out, int out_size, void* d_ws, size_t ws_size,
                              hipStream_t stream) {
  (void)in_sizes; (void)n_in; (void)out_size; (void)ws_size;
  const float* query = (const float*)d_in[0];
  const float* key   = (const float*)d_in[1];
  const float* value = (const float*)d_in[2];
  const int*   mask  = (const int*)d_in[3];
  const float* sem   = (const float*)d_in[4];
  const float* Wq = (const float*)d_in[5];
  const float* bq = (const float*)d_in[6];
  const float* Wk = (const float*)d_in[7];
  const float* bk = (const float*)d_in[8];
  const float* Wv = (const float*)d_in[9];
  const float* bv = (const float*)d_in[10];
  const float* Wo = (const float*)d_in[11];
  const float* bo = (const float*)d_in[12];
  const float* Wt1 = (const float*)d_in[13];
  const float* bt1 = (const float*)d_in[14];
  const float* Wt2 = (const float*)d_in[15];
  const float* bt2 = (const float*)d_in[16];
  float* out = (float*)d_out;

  char* base = (char*)d_ws;
  size_t off = 0;
  auto carve = [&](size_t bytes) -> void* {
    void* p = base + off;
    off += (bytes + 255) & ~(size_t)255;
    return p;
  };
  const int M = B_ * S_;  // 4096 tokens
  bf16* Wq_bf = (bf16*)carve((size_t)D_ * D_ * 2);
  bf16* Wk_bf = (bf16*)carve((size_t)D_ * D_ * 2);
  bf16* Wv_bf = (bf16*)carve((size_t)D_ * D_ * 2);
  bf16* Wo_bf = (bf16*)carve((size_t)D_ * D_ * 2);
  bf16* Xq_bf = (bf16*)carve((size_t)M * D_ * 2);
  bf16* Xk_bf = (bf16*)carve((size_t)M * D_ * 2);
  bf16* Xv_bf = (bf16*)carve((size_t)M * D_ * 2);
  bf16* Qh    = (bf16*)carve((size_t)M * D_ * 2);
  bf16* Kh    = (bf16*)carve((size_t)M * D_ * 2);
  bf16* Vt    = (bf16*)carve((size_t)M * D_ * 2);
  bf16* Sn_bf = (bf16*)carve((size_t)B_ * S_ * DSEM_ * 2);
  bf16* Ctx   = (bf16*)carve((size_t)M * D_ * 2);
  bf16* biasT = (bf16*)carve((size_t)B_ * S_ * S_ * 2);  // 16 MB, k-major
  float* qmean   = (float*)carve((size_t)B_ * D_ * 4);
  float* invtemp = (float*)carve(B_ * 4);

  // 1) f32 -> bf16 conversions
  k_convert<<<1024, 256, 0, stream>>>(Wq, Wq_bf, D_ * D_);
  k_convert<<<1024, 256, 0, stream>>>(Wk, Wk_bf, D_ * D_);
  k_convert<<<1024, 256, 0, stream>>>(Wv, Wv_bf, D_ * D_);
  k_convert<<<1024, 256, 0, stream>>>(Wo, Wo_bf, D_ * D_);
  k_convert<<<2048, 256, 0, stream>>>(query, Xq_bf, M * D_);
  k_convert<<<2048, 256, 0, stream>>>(key,   Xk_bf, M * D_);
  k_convert<<<2048, 256, 0, stream>>>(value, Xv_bf, M * D_);

  // 2) semantic feature normalization + one-shot semantic bias (head-independent)
  k_normalize<<<(B_ * S_) / 4, 128, 0, stream>>>(sem, Sn_bf);
  dim3 sgrid(S_ / 64, S_ / 16, B_);
  k_simbias<<<sgrid, 128, 0, stream>>>(Sn_bf, biasT);

  // 3) dynamic temperature
  k_qmean<<<(B_ * D_) / 256, 256, 0, stream>>>(query, qmean);
  k_temp<<<B_, 256, 0, stream>>>(qmean, Wt1, bt1, Wt2, bt2, invtemp);

  // 4) Q/K/V projections (WMMA bf16, 32x64 per wave)
  dim3 ggrid(M / 128, D_ / 64);
  k_gemm<<<ggrid, 128, 0, stream>>>(Xq_bf, Wq_bf, bq, Qh, nullptr, 0);
  k_gemm<<<ggrid, 128, 0, stream>>>(Xk_bf, Wk_bf, bk, Kh, nullptr, 1);
  k_gemm<<<ggrid, 128, 0, stream>>>(Xv_bf, Wv_bf, bv, Vt, nullptr, 2);

  // 5) fused flash attention with all constraint biases
  dim3 agrid(S_ / 64, H_, B_);
  k_attn<<<agrid, 128, 0, stream>>>(Qh, Kh, Vt, biasT, mask, invtemp, Ctx);

  // 6) output projection -> f32
  k_gemm<<<ggrid, 128, 0, stream>>>(Ctx, Wo_bf, bo, nullptr, out, 3);
}
